// Deform_align_309237645396
// MI455X (gfx1250) — matrix operand order repaired and checked by reading, not compiled, not run on hardware
//
#include <hip/hip_runtime.h>
#include <hip/hip_bf16.h>
#include <hip/hip_fp16.h>

typedef __attribute__((ext_vector_type(16))) _Float16 v16h;
typedef __attribute__((ext_vector_type(8)))  _Float16 v8h;
typedef __attribute__((ext_vector_type(8)))  float    v8f;

#define DEV __device__ __forceinline__

// Problem constants (from setup_inputs): B=4, C=64, H=W=160
constexpr int B_  = 4;
constexpr int C_  = 64;
constexpr int H_  = 160;
constexpr int W_  = 160;
constexpr int HW_ = H_ * W_;            // 25600
constexpr int M_TOT = B_ * HW_;         // 102400 spatial positions
constexpr int M_TILES = M_TOT / 16;     // 6400

DEV v8h ld8(const _Float16* p) {
    return *reinterpret_cast<const v8h*>(p);
}

// A-fragment (16x32 f16, wave32): lane = row(lane&15); ksel = lane>>4
// elems 0..7  = K (ksel*8 + e), elems 8..15 = K (16 + ksel*8 + e)
DEV v16h mkfragA(const _Float16* row_kbase, int ksel) {
    const _Float16* p = row_kbase + ksel * 8;
    v8h lo = ld8(p);
    v8h hi = ld8(p + 16);
    v16h a;
#pragma unroll
    for (int i = 0; i < 8; ++i) { a[i] = lo[i]; a[8 + i] = hi[i]; }
    return a;
}

// B-fragment (32x16 f16, wave32): lane = col(lane&15); elems 0..7 = K (ksel*8 + e), elems 8..15 = K (16 + ksel*8 + e)
DEV v16h mkfragB(const _Float16* col_kbase, int ksel) {
    const _Float16* p = col_kbase + ksel * 8;
    v8h lo = ld8(p);
    v8h hi = ld8(p + 16);
    v16h b;
#pragma unroll
    for (int i = 0; i < 8; ++i) { b[i] = lo[i]; b[8 + i] = hi[i]; }
    return b;
}

DEV v8f wmma_f16(v16h a, v16h b, v8f c) {
    return __builtin_amdgcn_wmma_f32_16x16x32_f16(
        /*neg_a=*/false, a, /*neg_b=*/false, b,
        /*c_mod=*/(short)0, c, /*reuse_a=*/false, /*reuse_b=*/false);
}

DEV int clampi(int v, int lo, int hi) { return v < lo ? lo : (v > hi ? hi : v); }

// ---------------------------------------------------------------------------
// Layout-conversion kernels (cheap, bandwidth-trivial vs 23.3 TB/s)
// ---------------------------------------------------------------------------

// NCHW f32 (aux, ref) -> NHWC f16 cat[128] and NHWC f16 aux[64]
__global__ void pack_act(const float* __restrict__ aux, const float* __restrict__ ref,
                         _Float16* __restrict__ cat16, _Float16* __restrict__ aux16) {
    int idx = blockIdx.x * blockDim.x + threadIdx.x;   // over B*C*HW (NCHW linear)
    if (idx >= B_ * C_ * HW_) return;
    int hw = idx % HW_;
    int c  = (idx / HW_) % C_;
    int b  = idx / (HW_ * C_);
    float va = aux[idx];
    float vr = ref[idx];
    int m = b * HW_ + hw;
    aux16[(size_t)m * 64 + c]        = (_Float16)va;
    cat16[(size_t)m * 128 + c]       = (_Float16)va;
    cat16[(size_t)m * 128 + 64 + c]  = (_Float16)vr;
}

// 1x1 weights [O][K] f32 -> [Opad][K] f16 (zero-pad rows O..Opad-1)
__global__ void pack_w1(const float* __restrict__ w, _Float16* __restrict__ wt,
                        int O, int K, int Opad) {
    int idx = blockIdx.x * blockDim.x + threadIdx.x;
    if (idx >= Opad * K) return;
    int o = idx / K;
    int k = idx % K;
    wt[idx] = (o < O) ? (_Float16)w[o * K + k] : (_Float16)0.0f;
}

// 3x3 weights OIHW [O][Ci][3][3] f32 -> [O][tap][Ci] f16   (tap = ky*3+kx)
__global__ void pack_w3(const float* __restrict__ w, _Float16* __restrict__ wt, int O, int Ci) {
    int idx = blockIdx.x * blockDim.x + threadIdx.x;
    if (idx >= O * Ci * 9) return;
    int ci = idx % Ci;
    int t  = (idx / Ci) % 9;
    int o  = idx / (Ci * 9);
    wt[idx] = (_Float16)w[(size_t)(o * Ci + ci) * 9 + t];
}

// Zero-fill a small f16 region (zero-pad row for out-of-bounds conv taps)
__global__ void zero_fill(_Float16* __restrict__ p, int n) {
    int idx = blockIdx.x * blockDim.x + threadIdx.x;
    if (idx < n) p[idx] = (_Float16)0.0f;
}

// ---------------------------------------------------------------------------
// 1x1 conv = plain GEMM: M=102400, K=KDIM, N=64 (or 18 padded to 32)
// 8 waves/block; each wave one 16x16 tile. NT n-tiles per m-tile.
// ---------------------------------------------------------------------------
template <int KDIM, int NT, bool LRELU, bool RESID, bool OUTF32>
__global__ void conv1x1_wmma(const _Float16* __restrict__ A,
                             const _Float16* __restrict__ Wt,   // [Opad][KDIM] f16
                             const float* __restrict__ bias,
                             const _Float16* __restrict__ resid, // NHWC 64ch f16
                             _Float16* __restrict__ out16,
                             float* __restrict__ out32,
                             int ldo, int choff, int nch) {
    constexpr int MPB = 8 / NT;
    int wave = threadIdx.x >> 5;
    int lane = threadIdx.x & 31;
    int mt = blockIdx.x * MPB + (wave / NT);
    int nt = wave % NT;

    int rc   = lane & 15;      // A-row / B-col within tile
    int ksel = lane >> 4;

    const _Float16* arow = A  + (size_t)(mt * 16 + rc) * KDIM;
    const _Float16* bcol = Wt + (size_t)(nt * 16 + rc) * KDIM;

    v8f acc = {};
#pragma unroll
    for (int kb = 0; kb < KDIM; kb += 32) {
        v16h a = mkfragA(arow + kb, ksel);
        v16h b = mkfragB(bcol + kb, ksel);
        acc = wmma_f16(a, b, acc);
    }

    int n = nt * 16 + rc;
    float bv = (n < nch) ? bias[n] : 0.0f;

#pragma unroll
    for (int r = 0; r < 8; ++r) {
        int m = mt * 16 + ksel * 8 + r;
        float v = acc[r] + bv;
        if (LRELU) v = (v >= 0.0f) ? v : 0.1f * v;
        if constexpr (RESID) v += (float)resid[(size_t)m * 64 + n];
        if constexpr (OUTF32) {
            if (n < nch) out32[(size_t)m * ldo + n] = v;
        } else {
            out16[(size_t)m * ldo + choff + n] = (_Float16)v;
        }
    }
}

// ---------------------------------------------------------------------------
// 3x3 dilated conv as implicit GEMM: K = 9 taps x 64ch, lrelu epilogue.
// W=160 is a multiple of 16 so each 16-row M-tile stays inside one image row.
// Out-of-bounds taps read a zero-pad row via branch-free pointer select
// (v_cndmask on the address, unconditional b128 loads -> no EXEC branches).
// ---------------------------------------------------------------------------
__global__ void conv3x3_wmma(const _Float16* __restrict__ A,   // NHWC 64ch f16
                             const _Float16* __restrict__ Wt,  // [64][9*64] f16
                             const float* __restrict__ bias,
                             const _Float16* __restrict__ zpad, // >=64 zero f16
                             _Float16* __restrict__ out,       // NHWC ldo ch f16
                             int ldo, int choff, int dil) {
    int wave = threadIdx.x >> 5;
    int lane = threadIdx.x & 31;
    int mt = blockIdx.x * 2 + (wave >> 2);   // NT=4 -> 2 m-tiles per block
    int nt = wave & 3;

    int rc   = lane & 15;
    int ksel = lane >> 4;

    int p = mt * 16 + rc;          // this lane's A-row spatial position
    int b = p / HW_;
    int hw = p % HW_;
    int h = hw / W_;
    int w = hw % W_;

    const _Float16* bcol = Wt + (size_t)(nt * 16 + rc) * 576;

    v8f acc = {};
#pragma unroll
    for (int t = 0; t < 9; ++t) {
        int ky = t / 3 - 1, kx = t % 3 - 1;
        int hh = h + ky * dil;
        int wp = w + kx * dil;
        bool valid = (hh >= 0) & (hh < H_) & (wp >= 0) & (wp < W_);
        const _Float16* real =
            A + (size_t)(b * HW_ + clampi(hh, 0, H_ - 1) * W_ + clampi(wp, 0, W_ - 1)) * 64;
        const _Float16* arow = valid ? real : zpad;   // branch-free select
#pragma unroll
        for (int s = 0; s < 2; ++s) {
            v16h a  = mkfragA(arow + s * 32, ksel);
            v16h bf = mkfragB(bcol + t * 64 + s * 32, ksel);
            acc = wmma_f16(a, bf, acc);
        }
    }

    int n = nt * 16 + rc;
    float bv = bias[n];
#pragma unroll
    for (int r = 0; r < 8; ++r) {
        int m = mt * 16 + ksel * 8 + r;
        float v = acc[r] + bv;
        v = (v >= 0.0f) ? v : 0.1f * v;
        out[(size_t)m * ldo + choff + n] = (_Float16)v;
    }
}

// ---------------------------------------------------------------------------
// Deformable 3x3 conv (torchvision offset layout): bilinear gather from NHWC
// f16 aux, packed-f16 blend (v_pk_fma_f16), K = 9x64 implicit GEMM,
// NCHW f32 output via aligned float4 stores.
// ---------------------------------------------------------------------------
__global__ void deform_wmma(const _Float16* __restrict__ aux16, // NHWC 64ch f16
                            const float* __restrict__ off,      // [M][18] f32 (y,x per tap)
                            const _Float16* __restrict__ dwt,   // [64][9*64] f16
                            const float* __restrict__ dbias,
                            float* __restrict__ out) {          // NCHW f32
    int wave = threadIdx.x >> 5;
    int lane = threadIdx.x & 31;
    int mt = blockIdx.x * 2 + (wave >> 2);
    int nt = wave & 3;

    int rc   = lane & 15;
    int ksel = lane >> 4;

    int p = mt * 16 + rc;
    int b = p / HW_;
    int hw = p % HW_;
    int h = hw / W_;
    int w = hw % W_;

    const _Float16* bcol = dwt + (size_t)(nt * 16 + rc) * 576;
    const _Float16* img  = aux16 + (size_t)b * HW_ * 64;

    v8f acc = {};
#pragma unroll
    for (int t = 0; t < 9; ++t) {
        int ky = t / 3, kx = t % 3;
        float dy = off[(size_t)p * 18 + 2 * t + 0];
        float dx = off[(size_t)p * 18 + 2 * t + 1];
        float py = (float)(h - 1 + ky) + dy;
        float px = (float)(w - 1 + kx) + dx;
        float fy = floorf(py), fx = floorf(px);
        float wy = py - fy,    wx = px - fx;
        int y0 = (int)fy, x0 = (int)fx;
        int y1 = y0 + 1,  x1 = x0 + 1;

        float v00 = ((y0 >= 0) & (y0 < H_) & (x0 >= 0) & (x0 < W_)) ? 1.0f : 0.0f;
        float v01 = ((y0 >= 0) & (y0 < H_) & (x1 >= 0) & (x1 < W_)) ? 1.0f : 0.0f;
        float v10 = ((y1 >= 0) & (y1 < H_) & (x0 >= 0) & (x0 < W_)) ? 1.0f : 0.0f;
        float v11 = ((y1 >= 0) & (y1 < H_) & (x1 >= 0) & (x1 < W_)) ? 1.0f : 0.0f;

        // Bilinear corner weights in f16 (accumulation precision recovered in
        // the f32 WMMA accumulator).
        _Float16 w00 = (_Float16)((1.0f - wy) * (1.0f - wx) * v00);
        _Float16 w01 = (_Float16)((1.0f - wy) * wx          * v01);
        _Float16 w10 = (_Float16)(wy          * (1.0f - wx) * v10);
        _Float16 w11 = (_Float16)(wy          * wx          * v11);

        int y0c = clampi(y0, 0, H_ - 1), y1c = clampi(y1, 0, H_ - 1);
        int x0c = clampi(x0, 0, W_ - 1), x1c = clampi(x1, 0, W_ - 1);
        const _Float16* p00 = img + (size_t)(y0c * W_ + x0c) * 64;
        const _Float16* p01 = img + (size_t)(y0c * W_ + x1c) * 64;
        const _Float16* p10 = img + (size_t)(y1c * W_ + x0c) * 64;
        const _Float16* p11 = img + (size_t)(y1c * W_ + x1c) * 64;

#pragma unroll
        for (int s = 0; s < 2; ++s) {
            int cb = s * 32 + ksel * 8;
            v16h a;
#pragma unroll
            for (int half = 0; half < 2; ++half) {
                int co = cb + half * 16;
                v8h q00 = ld8(p00 + co), q01 = ld8(p01 + co);
                v8h q10 = ld8(p10 + co), q11 = ld8(p11 + co);
                // packed f16 blend: lowers to v_pk_mul/fma_f16 (2 elems/op)
                v8h blend = q00 * w00 + q01 * w01 + q10 * w10 + q11 * w11;
#pragma unroll
                for (int e = 0; e < 8; ++e) a[half * 8 + e] = blend[e];
            }
            v16h bf = mkfragB(bcol + t * 64 + s * 32, ksel);
            acc = wmma_f16(a, bf, acc);
        }
    }

    int o = nt * 16 + rc;
    float bv = dbias[o];

    // Tile origin (tile is contiguous along w within one image row)
    int p0  = mt * 16;
    int b0  = p0 / HW_;
    int hw0 = p0 % HW_;
    size_t base = ((size_t)(b0 * 64 + o)) * HW_ + hw0 + ksel * 8;
    float4 o0 = make_float4(acc[0] + bv, acc[1] + bv, acc[2] + bv, acc[3] + bv);
    float4 o1 = make_float4(acc[4] + bv, acc[5] + bv, acc[6] + bv, acc[7] + bv);
    *reinterpret_cast<float4*>(out + base)     = o0;
    *reinterpret_cast<float4*>(out + base + 4) = o1;
}

// ---------------------------------------------------------------------------
// Host-side orchestration
// ---------------------------------------------------------------------------
extern "C" void kernel_launch(void* const* d_in, const int* in_sizes, int n_in,
                              void* d_out, int out_size, void* d_ws, size_t ws_size,
                              hipStream_t stream) {
    (void)in_sizes; (void)n_in; (void)out_size; (void)ws_size;

    const float* in_aux = (const float*)d_in[0];
    const float* in_ref = (const float*)d_in[1];
    const float* w01    = (const float*)d_in[2];
    const float* b01    = (const float*)d_in[3];
    const float* a1_w   = (const float*)d_in[4];
    const float* a1_b   = (const float*)d_in[5];
    const float* a2_w   = (const float*)d_in[6];
    const float* a2_b   = (const float*)d_in[7];
    const float* a3_w   = (const float*)d_in[8];
    const float* a3_b   = (const float*)d_in[9];
    const float* ar_w   = (const float*)d_in[10];
    const float* ar_b   = (const float*)d_in[11];
    const float* w02    = (const float*)d_in[12];
    const float* b02    = (const float*)d_in[13];
    const float* dw     = (const float*)d_in[14];
    const float* db     = (const float*)d_in[15];
    float* out = (float*)d_out;

    char* ws = (char*)d_ws;
    size_t o = 0;
    auto take = [&](size_t bytes) { size_t r = o; o += (bytes + 255) & ~(size_t)255; return r; };

    _Float16* cat16 = (_Float16*)(ws + take((size_t)M_TOT * 128 * 2));
    _Float16* aux16 = (_Float16*)(ws + take((size_t)M_TOT * 64 * 2));
    _Float16* x0    = (_Float16*)(ws + take((size_t)M_TOT * 64 * 2));
    _Float16* r123  = (_Float16*)(ws + take((size_t)M_TOT * 192 * 2));
    _Float16* x1    = (_Float16*)(ws + take((size_t)M_TOT * 64 * 2));
    float*    offb  = (float*)   (ws + take((size_t)M_TOT * 18 * 4));
    _Float16* w01t  = (_Float16*)(ws + take((size_t)64 * 128 * 2));
    _Float16* art   = (_Float16*)(ws + take((size_t)64 * 192 * 2));
    _Float16* w02t  = (_Float16*)(ws + take((size_t)32 * 64 * 2));
    _Float16* a1t   = (_Float16*)(ws + take((size_t)64 * 576 * 2));
    _Float16* a2t   = (_Float16*)(ws + take((size_t)64 * 576 * 2));
    _Float16* a3t   = (_Float16*)(ws + take((size_t)64 * 576 * 2));
    _Float16* dwt   = (_Float16*)(ws + take((size_t)64 * 576 * 2));
    _Float16* zpad  = (_Float16*)(ws + take((size_t)64 * 2));

    // 1) layout conversions
    pack_act<<<(B_ * C_ * HW_ + 255) / 256, 256, 0, stream>>>(in_aux, in_ref, cat16, aux16);
    pack_w1<<<(64 * 128 + 255) / 256, 256, 0, stream>>>(w01, w01t, 64, 128, 64);
    pack_w1<<<(64 * 192 + 255) / 256, 256, 0, stream>>>(ar_w, art, 64, 192, 64);
    pack_w1<<<(32 * 64 + 255) / 256, 256, 0, stream>>>(w02, w02t, 18, 64, 32);
    pack_w3<<<(64 * 64 * 9 + 255) / 256, 256, 0, stream>>>(a1_w, a1t, 64, 64);
    pack_w3<<<(64 * 64 * 9 + 255) / 256, 256, 0, stream>>>(a2_w, a2t, 64, 64);
    pack_w3<<<(64 * 64 * 9 + 255) / 256, 256, 0, stream>>>(a3_w, a3t, 64, 64);
    pack_w3<<<(64 * 64 * 9 + 255) / 256, 256, 0, stream>>>(dw, dwt, 64, 64);
    zero_fill<<<1, 64, 0, stream>>>(zpad, 64);

    // 2) conv01: [M,128]x[128,64] + lrelu  -> x0 (NHWC f16)
    conv1x1_wmma<128, 4, true, false, false><<<M_TILES / 2, 256, 0, stream>>>(
        cat16, w01t, b01, nullptr, x0, nullptr, 64, 0, 64);

    // 3) res-ASPP branches: three dilated 3x3 convs + lrelu -> r123 (NHWC 192ch)
    conv3x3_wmma<<<M_TILES / 2, 256, 0, stream>>>(x0, a1t, a1_b, zpad, r123, 192, 0,   1);
    conv3x3_wmma<<<M_TILES / 2, 256, 0, stream>>>(x0, a2t, a2_b, zpad, r123, 192, 64,  2);
    conv3x3_wmma<<<M_TILES / 2, 256, 0, stream>>>(x0, a3t, a3_b, zpad, r123, 192, 128, 4);

    // 4) fuse: [M,192]x[192,64] + residual(x0) -> x1
    conv1x1_wmma<192, 4, false, true, false><<<M_TILES / 2, 256, 0, stream>>>(
        r123, art, ar_b, x0, x1, nullptr, 64, 0, 64);

    // 5) conv02: [M,64]x[64,18] -> offsets f32
    conv1x1_wmma<64, 2, false, false, true><<<M_TILES / 4, 256, 0, stream>>>(
        x1, w02t, b02, nullptr, nullptr, offb, 18, 0, 18);

    // 6) deformable conv on aux -> NCHW f32 output
    deform_wmma<<<M_TILES / 2, 256, 0, stream>>>(aux16, offb, dwt, db, out);
}
